// SimpleMDNetNew_47313359732776
// MI455X (gfx1250) — matrix operand order repaired and checked by reading, not compiled, not run on hardware
//
#include <hip/hip_runtime.h>
#include <math.h>

typedef __attribute__((ext_vector_type(16))) __bf16 v16bf;
typedef __attribute__((ext_vector_type(8)))  float  v8f;

#define NNODES 50000
#define NEDGES 800000
#define DIM    128
#define ASTRIDE 130   // padded LDS row stride in shorts (65 dwords -> conflict-free)

__device__ __forceinline__ unsigned short f32_to_bf16(float f) {
  unsigned u = __float_as_uint(f);
  u += 0x7FFFu + ((u >> 16) & 1u);   // round-to-nearest-even
  return (unsigned short)(u >> 16);
}

__global__ void zero_kernel(float* __restrict__ p, int n) {
  int t = blockIdx.x * blockDim.x + threadIdx.x;
  if (t < n) p[t] = 0.f;
}

__global__ void cvt_w_kernel(const float* __restrict__ W, unsigned short* __restrict__ Wb, int n) {
  int t = blockIdx.x * blockDim.x + threadIdx.x;
  if (t < n) Wb[t] = f32_to_bf16(W[t]);
}

__global__ void degree_kernel(const int* __restrict__ dst, const int* __restrict__ src,
                              float* __restrict__ deg_out, float* __restrict__ deg_in, int E) {
  int t = blockIdx.x * blockDim.x + threadIdx.x;
  if (t < E) {
    atomicAdd(&deg_out[src[t]], 1.f);
    atomicAdd(&deg_in[dst[t]], 1.f);
  }
}

__global__ void norm_kernel(float* __restrict__ deg, int n) {
  int t = blockIdx.x * blockDim.x + threadIdx.x;
  if (t < n) deg[t] = rsqrtf(fmaxf(deg[t], 1.f));
}

// encoder layer 0: [N,3] @ [3,128] + b, leaky_relu(0.2)
__global__ void enc0_kernel(const float* __restrict__ pos, const float* __restrict__ W,
                            const float* __restrict__ b, float* __restrict__ out, int n) {
  int t = blockIdx.x * blockDim.x + threadIdx.x;
  if (t >= n * DIM) return;
  int node = t >> 7, c = t & 127;
  float x0 = pos[node * 3 + 0], x1 = pos[node * 3 + 1], x2 = pos[node * 3 + 2];
  float v = fmaf(x0, W[c], fmaf(x1, W[DIM + c], fmaf(x2, W[2 * DIM + c], b[c])));
  out[t] = v > 0.f ? v : 0.2f * v;
}

// per edge: agg[dst] += x[src] * norm_src[src]  (one wave32 per edge, float4 per lane)
__global__ void gather_scatter_kernel(const float* __restrict__ x, const int* __restrict__ src,
                                      const int* __restrict__ dst, const float* __restrict__ nsrc,
                                      float* __restrict__ agg, int E) {
  int t = blockIdx.x * blockDim.x + threadIdx.x;
  int e = t >> 5;
  if (e >= E) return;
  int lane = t & 31;
  int s = src[e], d = dst[e];
  float sc = nsrc[s];
  float4 v = *reinterpret_cast<const float4*>(&x[(size_t)s * DIM + lane * 4]);
  float* ap = &agg[(size_t)d * DIM + lane * 4];
  atomicAdd(ap + 0, v.x * sc);
  atomicAdd(ap + 1, v.y * sc);
  atomicAdd(ap + 2, v.z * sc);
  atomicAdd(ap + 3, v.w * sc);
}

// Y[16x128 block] = act( (rowscale ? diag(rs)*X : X) @ W + b ), W bf16 [128,128] row-major.
// 256 threads = 8 waves; wave w computes cols [16w,16w+16) via 4x v_wmma_f32_16x16x32_bf16.
// ACT: 0 none, 1 leaky(0.2), 2 tanh, 3 silu
template <int ACT>
__global__ void gemm_bf16_kernel(const float* __restrict__ X, const unsigned short* __restrict__ Wb,
                                 const float* __restrict__ bias, const float* __restrict__ rowscale,
                                 float* __restrict__ Y) {
  __shared__ unsigned short As[16 * ASTRIDE];
  const int tid = threadIdx.x;
  const int rowbase = blockIdx.x * 16;

  // stage 16x128 A panel to LDS as bf16 (row-scaled)
#pragma unroll
  for (int i = 0; i < 8; ++i) {
    int e = tid + i * 256;
    int r = e >> 7, c = e & 127;
    float v = X[(size_t)(rowbase + r) * DIM + c];
    if (rowscale) v *= rowscale[rowbase + r];
    As[r * ASTRIDE + c] = f32_to_bf16(v);
  }
  __syncthreads();

  const int wave = tid >> 5;
  const int lane = tid & 31;
  const int half = lane >> 4;
  const int m    = lane & 15;
  const int nbase = wave * 16;
  const int ncol  = nbase + m;      // C/D: lane -> column

  v8f acc;
  {
    float bv = bias[ncol];
#pragma unroll
    for (int r = 0; r < 8; ++r) acc[r] = bv;
  }

  const int krow_lane = m + 16 * half;  // unused; B: lane -> column ncol, K by element

#pragma unroll
  for (int kc = 0; kc < DIM; kc += 32) {
    union { unsigned u[8]; v16bf v; } A;
#pragma unroll
    for (int vv = 0; vv < 4; ++vv) {            // VGPR 0..3: K = 8*half + 2v (+1)
      int k = kc + 8 * half + 2 * vv;
      A.u[vv] = *reinterpret_cast<const unsigned*>(&As[m * ASTRIDE + k]);
    }
#pragma unroll
    for (int vv = 0; vv < 4; ++vv) {            // VGPR 4..7: K = 16 + 8*half + 2v (+1)
      int k = kc + 16 + 8 * half + 2 * vv;
      A.u[4 + vv] = *reinterpret_cast<const unsigned*>(&As[m * ASTRIDE + k]);
    }
    union { unsigned u[8]; v16bf v; } B;
#pragma unroll
    for (int vv = 0; vv < 8; ++vv) {            // VGPR v: column ncol, K same as A (table order)
      int k = kc + 8 * half + 2 * vv + (vv >= 4 ? 8 : 0);
      B.u[vv] = (unsigned)Wb[(size_t)k * DIM + ncol] | ((unsigned)Wb[(size_t)(k + 1) * DIM + ncol] << 16);
    }
    acc = __builtin_amdgcn_wmma_f32_16x16x32_bf16(false, A.v, false, B.v,
                                                  (short)0, acc, false, false);
  }

#pragma unroll
  for (int r = 0; r < 8; ++r) {
    float v = acc[r];
    if (ACT == 1)      v = v > 0.f ? v : 0.2f * v;
    else if (ACT == 2) v = tanhf(v);
    else if (ACT == 3) v = v / (1.f + expf(-v));
    int row = rowbase + r + 8 * half;           // C/D: VGPR r -> M = r + 8*half
    Y[(size_t)row * DIM + ncol] = v;
  }
}

// decoder last layer: [N,128] @ [128,3] + b
__global__ void dec_last_kernel(const float* __restrict__ h, const float* __restrict__ W,
                                const float* __restrict__ b, float* __restrict__ out, int n) {
  int t = blockIdx.x * blockDim.x + threadIdx.x;
  if (t >= n * 3) return;
  int node = t / 3, c = t % 3;
  const float* hr = &h[(size_t)node * DIM];
  float acc = b[c];
#pragma unroll 8
  for (int k = 0; k < DIM; ++k) acc = fmaf(hr[k], W[k * 3 + c], acc);
  out[t] = acc;
}

static inline void launch_gemm(int act, const float* X, const unsigned short* Wb,
                               const float* b, const float* rs, float* Y, hipStream_t s) {
  dim3 g(NNODES / 16), blk(256);
  switch (act) {
    case 0: gemm_bf16_kernel<0><<<g, blk, 0, s>>>(X, Wb, b, rs, Y); break;
    case 1: gemm_bf16_kernel<1><<<g, blk, 0, s>>>(X, Wb, b, rs, Y); break;
    case 2: gemm_bf16_kernel<2><<<g, blk, 0, s>>>(X, Wb, b, rs, Y); break;
    default: gemm_bf16_kernel<3><<<g, blk, 0, s>>>(X, Wb, b, rs, Y); break;
  }
}

extern "C" void kernel_launch(void* const* d_in, const int* in_sizes, int n_in,
                              void* d_out, int out_size, void* d_ws, size_t ws_size,
                              hipStream_t stream) {
  (void)in_sizes; (void)n_in; (void)out_size; (void)ws_size;

  // ---- inputs (dict order, leaves flattened) ----
  const float* pos  = (const float*)d_in[0];
  const int*   eidx = (const int*)d_in[1];
  const int*   dst  = eidx;              // edge_index[0] = center (dst)
  const int*   src  = eidx + NEDGES;     // edge_index[1] = neighbor (src)

  const float *encW[4], *encB[4], *convW[6], *convB[6], *decW[4], *decB[4];
  for (int i = 0; i < 4; ++i) { encW[i]  = (const float*)d_in[2  + 2 * i]; encB[i]  = (const float*)d_in[3  + 2 * i]; }
  for (int i = 0; i < 6; ++i) { convW[i] = (const float*)d_in[10 + 2 * i]; convB[i] = (const float*)d_in[11 + 2 * i]; }
  for (int i = 0; i < 4; ++i) { decW[i]  = (const float*)d_in[22 + 2 * i]; decB[i]  = (const float*)d_in[23 + 2 * i]; }

  // ---- workspace layout ----
  float* ws  = (float*)d_ws;
  float* hA  = ws;                                  // [N,128]
  float* hB  = hA  + (size_t)NNODES * DIM;          // [N,128]
  float* agg = hB  + (size_t)NNODES * DIM;          // [N,128]
  float* deg = agg + (size_t)NNODES * DIM;          // [2N]: deg_out | deg_in (-> norms in place)
  unsigned short* wbf = (unsigned short*)(deg + 2 * NNODES);  // 12 x 128*128 bf16
  const int WSZ = DIM * DIM;

  // ---- degrees + norms ----
  zero_kernel<<<(2 * NNODES + 255) / 256, 256, 0, stream>>>(deg, 2 * NNODES);
  degree_kernel<<<(NEDGES + 255) / 256, 256, 0, stream>>>(dst, src, deg, deg + NNODES, NEDGES);
  norm_kernel<<<(2 * NNODES + 255) / 256, 256, 0, stream>>>(deg, 2 * NNODES);
  const float* norm_src = deg;            // deg_out^-1/2
  const float* norm_dst = deg + NNODES;   // deg_in^-1/2

  // ---- convert 128x128 weights to bf16: enc1..3 (slots 0-2), conv0..5 (3-8), dec0..2 (9-11) ----
  for (int i = 0; i < 3; ++i)
    cvt_w_kernel<<<(WSZ + 255) / 256, 256, 0, stream>>>(encW[i + 1], wbf + (size_t)i * WSZ, WSZ);
  for (int i = 0; i < 6; ++i)
    cvt_w_kernel<<<(WSZ + 255) / 256, 256, 0, stream>>>(convW[i], wbf + (size_t)(3 + i) * WSZ, WSZ);
  for (int i = 0; i < 3; ++i)
    cvt_w_kernel<<<(WSZ + 255) / 256, 256, 0, stream>>>(decW[i], wbf + (size_t)(9 + i) * WSZ, WSZ);

  // ---- encoder ----
  enc0_kernel<<<(NNODES * DIM + 255) / 256, 256, 0, stream>>>(pos, encW[0], encB[0], hA, NNODES);
  launch_gemm(1, hA, wbf + 0 * (size_t)WSZ, encB[1], nullptr, hB, stream);
  launch_gemm(1, hB, wbf + 1 * (size_t)WSZ, encB[2], nullptr, hA, stream);
  launch_gemm(1, hA, wbf + 2 * (size_t)WSZ, encB[3], nullptr, hB, stream);

  // ---- 6 GraphConv layers (acts: none, tanh, silu, silu, tanh, none) ----
  const int acts[6] = {0, 2, 3, 3, 2, 0};
  float* cur = hB;
  float* other = hA;
  for (int l = 0; l < 6; ++l) {
    zero_kernel<<<(NNODES * DIM + 255) / 256, 256, 0, stream>>>(agg, NNODES * DIM);
    gather_scatter_kernel<<<((size_t)NEDGES * 32 + 255) / 256, 256, 0, stream>>>(
        cur, src, dst, norm_src, agg, NEDGES);
    launch_gemm(acts[l], agg, wbf + (size_t)(3 + l) * WSZ, convB[l], norm_dst, other, stream);
    float* t = cur; cur = other; other = t;
  }
  // cur == hB after 6 swaps

  // ---- decoder ----
  launch_gemm(1, cur,   wbf + 9  * (size_t)WSZ, decB[0], nullptr, other, stream);
  launch_gemm(1, other, wbf + 10 * (size_t)WSZ, decB[1], nullptr, cur,   stream);
  launch_gemm(1, cur,   wbf + 11 * (size_t)WSZ, decB[2], nullptr, other, stream);
  dec_last_kernel<<<(NNODES * 3 + 255) / 256, 256, 0, stream>>>(other, decW[3], decB[3],
                                                                (float*)d_out, NNODES);
}
